// MultiHeadAttentionManual_50027779064502
// MI455X (gfx1250) — matrix operand order repaired and checked by reading, not compiled, not run on hardware
//
#include <hip/hip_runtime.h>

// ---------------------------------------------------------------------------
// MultiHeadAttention forward for MI455X (gfx1250), wave32 + WMMA bf16 + TDM.
//   D_MODEL=1024, N_HEADS=16, D_K=64, B=2, S=2048.
// d_out = [output (2*2048*1024 f32)] ++ [attn_weights (2*16*2048*2048 f32)]
// Roofline: 537 MB attn-weights write @ 23.3 TB/s ~= 23 us floor; K/V (16 MB)
// live in the 192 MB L2, so scores are recomputed (3 passes) instead of
// spilling a 537 MB intermediate. All GEMMs are v_wmma_f32_16x16x32_bf16.
// Tile movement uses the Tensor Data Mover (tensor_load_to_lds) with double
// buffering: the DMA for tile i+1 overlaps the WMMAs on tile i, retired with
// s_wait_tensorcnt.
// ---------------------------------------------------------------------------

#define D_MODEL 1024
#define N_HEADS 16
#define D_K     64
#define BATCH   2
#define SEQ     2048

typedef __bf16 bf16_t;
typedef __attribute__((ext_vector_type(16))) __bf16 v16bf;
typedef __attribute__((ext_vector_type(8)))  __bf16 v8bf;
typedef __attribute__((ext_vector_type(8)))  float  v8f;
typedef unsigned int u32x4 __attribute__((ext_vector_type(4)));
typedef int          i32x4 __attribute__((ext_vector_type(4)));
typedef int          i32x8 __attribute__((ext_vector_type(8)));

#if __has_builtin(__builtin_amdgcn_tensor_load_to_lds)
#define HAVE_TDM 1
#else
#define HAVE_TDM 0
#endif

__device__ __forceinline__ bf16_t tobf(float f) {
  unsigned u = __builtin_bit_cast(unsigned, f);
  unsigned r = (u + 0x7fffu + ((u >> 16) & 1u)) >> 16;
  unsigned short s = (unsigned short)r;
  return __builtin_bit_cast(bf16_t, s);
}

__device__ __forceinline__ v8f zero8() {
  v8f z = {0.f, 0.f, 0.f, 0.f, 0.f, 0.f, 0.f, 0.f};
  return z;
}

__device__ __forceinline__ v8f wmma_bf16(v16bf a, v16bf b, v8f c) {
  return __builtin_amdgcn_wmma_f32_16x16x32_bf16(
      false, a, false, b, (short)0, c, false, false);
}

#if HAVE_TDM
// Low 32 bits of the generic address of an LDS object == LDS byte offset.
__device__ __forceinline__ unsigned lds_off_of(const void* p) {
  return (unsigned)(unsigned long long)(size_t)p;
}

// TDM: load a [tile_h x tile_w] bf16 tile (row stride row_stride_elems in
// memory) into LDS at lds_off, inserting pad_amt(+1) dwords of LDS padding
// every 2^(pad_int+1) dwords (row padding for bank-conflict-free WMMA frags).
__device__ __forceinline__ void tdm_load_tile_bf16(
    unsigned lds_off, const void* gaddr, unsigned tile_w, unsigned tile_h,
    unsigned row_stride_elems, unsigned pad_int, unsigned pad_amt) {
  unsigned long long ga = (unsigned long long)(size_t)gaddr;
  u32x4 g0;
  g0[0] = 1u;                                             // count=1 user D#
  g0[1] = lds_off;                                        // lds_addr
  g0[2] = (unsigned)ga;                                   // global_addr[31:0]
  g0[3] = (unsigned)((ga >> 32) & 0x01ffffffu) | (2u << 30);  // [56:32]|type=2
  i32x8 g1;
  g1[0] = (int)((1u << 16)            // data_size = 2 bytes
              | (1u << 20)            // pad_enable
              | (pad_int << 22)       // pad_interval
              | (pad_amt << 25));     // pad_amount
  g1[1] = (int)(tile_w << 16);        // tensor_dim0 = tile_w  (low 16)
  g1[2] = (int)(tile_h << 16);        // tensor_dim1 = tile_h  (low 16)
  g1[3] = (int)(tile_w << 16);        // tile_dim0
  g1[4] = (int)tile_h;                // tile_dim1 (tile_dim2 = 0)
  g1[5] = (int)row_stride_elems;      // tensor_dim0_stride (low 32)
  g1[6] = 0;
  g1[7] = 0;
  i32x4 gz = {0, 0, 0, 0};
#if defined(__clang_major__) && (__clang_major__ >= 23)
  i32x8 gz8 = {0, 0, 0, 0, 0, 0, 0, 0};
  __builtin_amdgcn_tensor_load_to_lds(g0, g1, gz, gz, gz8, 0);
#else
  __builtin_amdgcn_tensor_load_to_lds(g0, g1, gz, gz, 0);
#endif
}

template <int N>
__device__ __forceinline__ void tdm_wait() {
  __builtin_amdgcn_s_wait_tensorcnt(N);   // immediate constant required
}
#endif  // HAVE_TDM

// A-matrix fragment (16x32 bf16, MxK) from LDS tile stored [M][K] row-major.
__device__ __forceinline__ v16bf load_a_frag(const bf16_t* lds, int stride,
                                             int mbase, int lane) {
  const bf16_t* p = lds + (size_t)(mbase + (lane & 15)) * stride;
  const int khalf = (lane >> 4) * 16;
  v16bf a;
#pragma unroll
  for (int v = 0; v < 8; ++v) {
    int k = 2 * v + khalf;
    a[2 * v]     = p[k];
    a[2 * v + 1] = p[k + 1];
  }
  return a;
}

// B fragment (32x16 KxN) where B[k][n] is stored in LDS as tile[n][k].
__device__ __forceinline__ v16bf load_b_frag_nk(const bf16_t* lds, int stride,
                                                int nbase, int lane) {
  const bf16_t* p = lds + (size_t)(nbase + (lane & 15)) * stride + (lane >> 4) * 16;
  v16bf b;
#pragma unroll
  for (int v = 0; v < 16; ++v) b[v] = p[v];
  return b;
}

// B fragment (32x16 KxN) from LDS tile stored [K][N].
__device__ __forceinline__ v16bf load_b_frag_kn(const bf16_t* lds, int stride,
                                                int nbase, int lane) {
  const bf16_t* p = lds + (size_t)((lane >> 4) * 16) * stride + nbase + (lane & 15);
  v16bf b;
#pragma unroll
  for (int v = 0; v < 16; ++v) b[v] = p[(size_t)v * stride];
  return b;
}

// ---------------------------------------------------------------------------
__global__ __launch_bounds__(256) void cvt_f32_to_bf16(
    const float* __restrict__ in, bf16_t* __restrict__ out, int n) {
  const int i = (blockIdx.x * 256 + threadIdx.x) * 8;
  if (i + 8 <= n) {
    const float4 x0 = *(const float4*)(in + i);
    const float4 x1 = *(const float4*)(in + i + 4);
    v8bf o;
    o[0] = tobf(x0.x); o[1] = tobf(x0.y); o[2] = tobf(x0.z); o[3] = tobf(x0.w);
    o[4] = tobf(x1.x); o[5] = tobf(x1.y); o[6] = tobf(x1.z); o[7] = tobf(x1.w);
    *(v8bf*)(out + i) = o;
  }
}

// ---------------------------------------------------------------------------
// Y = X @ W^T + b, double-buffered TDM tiles, 64x64 block / 2x2 WMMA per wave.
// MODE 0: bf16 out in [B,H,S,64]; MODE 1: fp32 out flat [4096,1024].
// ---------------------------------------------------------------------------
template <int MODE>
__global__ __launch_bounds__(128) void gemm_xwT(
    const bf16_t* __restrict__ X, const bf16_t* __restrict__ W,
    const float* __restrict__ bias, void* __restrict__ out) {
  __shared__ bf16_t As[2][64][40];   // 32 elems + 4-dword pad per row
  __shared__ bf16_t Bs[2][64][40];   // W tile stored [n][k]

  const int tid  = threadIdx.x;
  const int lane = tid & 31;
  const int w    = tid >> 5;
  const int bx   = blockIdx.x;       // N/64
  const int by   = blockIdx.y;       // M/64
  const int m0   = (w >> 1) * 32;
  const int n0   = (w & 1) * 32;

  const bf16_t* xbase = X + (size_t)(by * 64) * D_MODEL;
  const bf16_t* wbase = W + (size_t)(bx * 64) * D_MODEL;

  v8f acc[2][2];
  acc[0][0] = zero8(); acc[0][1] = zero8();
  acc[1][0] = zero8(); acc[1][1] = zero8();

#if HAVE_TDM
  if (w == 0) {  // pad: every 16 dwords (code 3) insert 4 dwords (code 3)
    tdm_load_tile_bf16(lds_off_of(&As[0][0][0]), xbase, 32, 64, D_MODEL, 3, 3);
    tdm_load_tile_bf16(lds_off_of(&Bs[0][0][0]), wbase, 32, 64, D_MODEL, 3, 3);
  }
#endif

  const int NIT = D_MODEL / 32;
  for (int it = 0; it < NIT; ++it) {
    const int buf = it & 1;
    __syncthreads();
#if HAVE_TDM
    if (w == 0) {
      if (it + 1 < NIT) {  // prefetch next tiles, retire current pair
        tdm_load_tile_bf16(lds_off_of(&As[buf ^ 1][0][0]),
                           xbase + (it + 1) * 32, 32, 64, D_MODEL, 3, 3);
        tdm_load_tile_bf16(lds_off_of(&Bs[buf ^ 1][0][0]),
                           wbase + (it + 1) * 32, 32, 64, D_MODEL, 3, 3);
        tdm_wait<2>();
      } else {
        tdm_wait<0>();
      }
    }
#else
    {
      const int lrow = tid >> 1, lseg = (tid & 1) * 16;
      *(v16bf*)&As[buf][lrow][lseg] =
          *(const v16bf*)(xbase + (size_t)lrow * D_MODEL + it * 32 + lseg);
      *(v16bf*)&Bs[buf][lrow][lseg] =
          *(const v16bf*)(wbase + (size_t)lrow * D_MODEL + it * 32 + lseg);
    }
#endif
    __syncthreads();
    v16bf a0 = load_a_frag(&As[buf][0][0], 40, m0,      lane);
    v16bf a1 = load_a_frag(&As[buf][0][0], 40, m0 + 16, lane);
    v16bf b0 = load_b_frag_nk(&Bs[buf][0][0], 40, n0,      lane);
    v16bf b1 = load_b_frag_nk(&Bs[buf][0][0], 40, n0 + 16, lane);
    acc[0][0] = wmma_bf16(a0, b0, acc[0][0]);
    acc[0][1] = wmma_bf16(a0, b1, acc[0][1]);
    acc[1][0] = wmma_bf16(a1, b0, acc[1][0]);
    acc[1][1] = wmma_bf16(a1, b1, acc[1][1]);
  }

#pragma unroll
  for (int mt = 0; mt < 2; ++mt) {
#pragma unroll
    for (int nt = 0; nt < 2; ++nt) {
      const int n  = bx * 64 + n0 + nt * 16 + (lane & 15);
      const float bv = bias[n];
      const int mb = by * 64 + m0 + mt * 16 + (lane >> 4) * 8;
#pragma unroll
      for (int r = 0; r < 8; ++r) {
        const int m = mb + r;
        const float val = acc[mt][nt][r] + bv;
        if (MODE == 0) {
          const int b = m >> 11, s = m & (SEQ - 1);
          const int h = n >> 6, dk = n & (D_K - 1);
          ((bf16_t*)out)[(((size_t)(b * N_HEADS + h)) * SEQ + s) * D_K + dk] =
              tobf(val);
        } else {
          ((float*)out)[(size_t)m * D_MODEL + n] = val;
        }
      }
    }
  }
}

// ---------------------------------------------------------------------------
// Fused attention per (b,h, 64-query block); TDM double-buffered K/V tiles.
// pass 1: row max;  pass 2: exp-sum + unnormalized P@V (P relayout via LDS);
// pass 3: recompute + write normalized attn_weights fp32 exactly once.
// ---------------------------------------------------------------------------
__global__ __launch_bounds__(128) void attn_kernel(
    const bf16_t* __restrict__ Qp, const bf16_t* __restrict__ Kp,
    const bf16_t* __restrict__ Vp, float* __restrict__ attn_out,
    bf16_t* __restrict__ ctx_out) {
  __shared__ bf16_t Qs[64][72];
  __shared__ bf16_t Ks[2][64][72];
  __shared__ bf16_t Vs[2][64][72];
  __shared__ bf16_t Ps[4][16][72];
  __shared__ float  rowmax[64];
  __shared__ float  rowsum[64];

  const int tid  = threadIdx.x;
  const int lane = tid & 31;
  const int w    = tid >> 5;
  const int half = lane >> 4;
  const int nl   = lane & 15;
  const int qb   = blockIdx.x;
  const int bh   = blockIdx.y;
  const int bb   = bh >> 4, hh = bh & 15;
  const int q0   = qb * 64;
  const int m0   = w * 16;
  const float scale = 0.125f;  // 1/sqrt(64)
  const int NT = SEQ / 64;

  const bf16_t* Qh = Qp + (size_t)bh * SEQ * D_K;
  const bf16_t* Kh = Kp + (size_t)bh * SEQ * D_K;
  const bf16_t* Vh = Vp + (size_t)bh * SEQ * D_K;

  // Q tile (resident): [64,64] contiguous; pad every 32 dwords (code 4) by 4.
#if HAVE_TDM
  if (w == 0) {
    tdm_load_tile_bf16(lds_off_of(&Qs[0][0]), Qh + (size_t)q0 * D_K,
                       64, 64, 64, 4, 3);
    tdm_wait<0>();
  }
#else
  {
    const int row = tid >> 1, seg = (tid & 1) * 32;
    const bf16_t* src = Qh + (size_t)(q0 + row) * D_K + seg;
    *(v16bf*)&Qs[row][seg]      = *(const v16bf*)src;
    *(v16bf*)&Qs[row][seg + 16] = *(const v16bf*)(src + 16);
  }
#endif
  __syncthreads();

  const v16bf a0 = load_a_frag(&Qs[0][0],  72, m0, lane);
  const v16bf a1 = load_a_frag(&Qs[0][32], 72, m0, lane);

  // ---------------- pass 1: row max ----------------
  float rmax[8];
#pragma unroll
  for (int r = 0; r < 8; ++r) rmax[r] = -3.0e38f;

#if HAVE_TDM
  if (w == 0)
    tdm_load_tile_bf16(lds_off_of(&Ks[0][0][0]), Kh, 64, 64, 64, 4, 3);
#endif
  for (int kt = 0; kt < NT; ++kt) {
    const int buf = kt & 1;
    __syncthreads();
#if HAVE_TDM
    if (w == 0) {
      if (kt + 1 < NT) {
        tdm_load_tile_bf16(lds_off_of(&Ks[buf ^ 1][0][0]),
                           Kh + (size_t)(kt + 1) * 64 * D_K, 64, 64, 64, 4, 3);
        tdm_wait<1>();
      } else {
        tdm_wait<0>();
      }
    }
#else
    {
      const int row = tid >> 1, seg = (tid & 1) * 32;
      const bf16_t* src = Kh + (size_t)(kt * 64 + row) * D_K + seg;
      *(v16bf*)&Ks[buf][row][seg]      = *(const v16bf*)src;
      *(v16bf*)&Ks[buf][row][seg + 16] = *(const v16bf*)(src + 16);
    }
#endif
    __syncthreads();
#pragma unroll
    for (int nt = 0; nt < 4; ++nt) {
      v8f s = zero8();
      v16bf b0 = load_b_frag_nk(&Ks[buf][0][0],  72, nt * 16, lane);
      v16bf b1 = load_b_frag_nk(&Ks[buf][0][32], 72, nt * 16, lane);
      s = wmma_bf16(a0, b0, s);
      s = wmma_bf16(a1, b1, s);
#pragma unroll
      for (int r = 0; r < 8; ++r) rmax[r] = fmaxf(rmax[r], s[r] * scale);
    }
  }
#pragma unroll
  for (int r = 0; r < 8; ++r) {
    float v = rmax[r];
    v = fmaxf(v, __shfl_xor(v, 1, 32));
    v = fmaxf(v, __shfl_xor(v, 2, 32));
    v = fmaxf(v, __shfl_xor(v, 4, 32));
    v = fmaxf(v, __shfl_xor(v, 8, 32));
    if (nl == 0) rowmax[m0 + half * 8 + r] = v;
  }
  __syncthreads();

  float mrow[8];
#pragma unroll
  for (int r = 0; r < 8; ++r) mrow[r] = rowmax[m0 + half * 8 + r];

  // ---------------- pass 2: exp-sum + P@V ----------------
  float rsum[8] = {0.f, 0.f, 0.f, 0.f, 0.f, 0.f, 0.f, 0.f};
  v8f cacc[4];
#pragma unroll
  for (int nt = 0; nt < 4; ++nt) cacc[nt] = zero8();

#if HAVE_TDM
  if (w == 0) {
    tdm_load_tile_bf16(lds_off_of(&Ks[0][0][0]), Kh, 64, 64, 64, 4, 3);
    tdm_load_tile_bf16(lds_off_of(&Vs[0][0][0]), Vh, 64, 64, 64, 4, 3);
  }
#endif
  for (int kt = 0; kt < NT; ++kt) {
    const int buf = kt & 1;
    __syncthreads();
#if HAVE_TDM
    if (w == 0) {
      if (kt + 1 < NT) {
        tdm_load_tile_bf16(lds_off_of(&Ks[buf ^ 1][0][0]),
                           Kh + (size_t)(kt + 1) * 64 * D_K, 64, 64, 64, 4, 3);
        tdm_load_tile_bf16(lds_off_of(&Vs[buf ^ 1][0][0]),
                           Vh + (size_t)(kt + 1) * 64 * D_K, 64, 64, 64, 4, 3);
        tdm_wait<2>();
      } else {
        tdm_wait<0>();
      }
    }
#else
    {
      const int row = tid >> 1, seg = (tid & 1) * 32;
      const bf16_t* ksrc = Kh + (size_t)(kt * 64 + row) * D_K + seg;
      const bf16_t* vsrc = Vh + (size_t)(kt * 64 + row) * D_K + seg;
      *(v16bf*)&Ks[buf][row][seg]      = *(const v16bf*)ksrc;
      *(v16bf*)&Ks[buf][row][seg + 16] = *(const v16bf*)(ksrc + 16);
      *(v16bf*)&Vs[buf][row][seg]      = *(const v16bf*)vsrc;
      *(v16bf*)&Vs[buf][row][seg + 16] = *(const v16bf*)(vsrc + 16);
    }
#endif
    __syncthreads();
#pragma unroll
    for (int nt = 0; nt < 4; ++nt) {
      v8f s = zero8();
      v16bf b0 = load_b_frag_nk(&Ks[buf][0][0],  72, nt * 16, lane);
      v16bf b1 = load_b_frag_nk(&Ks[buf][0][32], 72, nt * 16, lane);
      s = wmma_bf16(a0, b0, s);
      s = wmma_bf16(a1, b1, s);
#pragma unroll
      for (int r = 0; r < 8; ++r) {
        const float p = __expf(s[r] * scale - mrow[r]);
        rsum[r] += p;
        Ps[w][half * 8 + r][nt * 16 + nl] = tobf(p);
      }
    }
    __syncthreads();
    const v16bf pa0 = load_a_frag(&Ps[w][0][0],  72, 0, lane);
    const v16bf pa1 = load_a_frag(&Ps[w][0][32], 72, 0, lane);
#pragma unroll
    for (int nt = 0; nt < 4; ++nt) {
      v16bf vb0 = load_b_frag_kn(&Vs[buf][0][0],  72, nt * 16, lane);
      v16bf vb1 = load_b_frag_kn(&Vs[buf][32][0], 72, nt * 16, lane);
      cacc[nt] = wmma_bf16(pa0, vb0, cacc[nt]);
      cacc[nt] = wmma_bf16(pa1, vb1, cacc[nt]);
    }
  }
#pragma unroll
  for (int r = 0; r < 8; ++r) {
    float v = rsum[r];
    v += __shfl_xor(v, 1, 32);
    v += __shfl_xor(v, 2, 32);
    v += __shfl_xor(v, 4, 32);
    v += __shfl_xor(v, 8, 32);
    if (nl == 0) rowsum[m0 + half * 8 + r] = v;
  }
  __syncthreads();

  float sinv[8];
#pragma unroll
  for (int r = 0; r < 8; ++r) sinv[r] = 1.0f / rowsum[m0 + half * 8 + r];

#pragma unroll
  for (int nt = 0; nt < 4; ++nt) {
#pragma unroll
    for (int r = 0; r < 8; ++r) {
      const int mloc = m0 + half * 8 + r;
      const float cv = cacc[nt][r] * sinv[r];
      ctx_out[(size_t)(bb * SEQ + q0 + mloc) * D_MODEL + hh * D_K + nt * 16 + nl] =
          tobf(cv);
    }
  }

  // ---------------- pass 3: normalized attn_weights ----------------
  float* abase = attn_out + ((size_t)bh * SEQ + q0) * SEQ;
#if HAVE_TDM
  if (w == 0)
    tdm_load_tile_bf16(lds_off_of(&Ks[0][0][0]), Kh, 64, 64, 64, 4, 3);
#endif
  for (int kt = 0; kt < NT; ++kt) {
    const int buf = kt & 1;
    __syncthreads();
#if HAVE_TDM
    if (w == 0) {
      if (kt + 1 < NT) {
        tdm_load_tile_bf16(lds_off_of(&Ks[buf ^ 1][0][0]),
                           Kh + (size_t)(kt + 1) * 64 * D_K, 64, 64, 64, 4, 3);
        tdm_wait<1>();
      } else {
        tdm_wait<0>();
      }
    }
#else
    {
      const int row = tid >> 1, seg = (tid & 1) * 32;
      const bf16_t* src = Kh + (size_t)(kt * 64 + row) * D_K + seg;
      *(v16bf*)&Ks[buf][row][seg]      = *(const v16bf*)src;
      *(v16bf*)&Ks[buf][row][seg + 16] = *(const v16bf*)(src + 16);
    }
#endif
    __syncthreads();
#pragma unroll
    for (int nt = 0; nt < 4; ++nt) {
      v8f s = zero8();
      v16bf b0 = load_b_frag_nk(&Ks[buf][0][0],  72, nt * 16, lane);
      v16bf b1 = load_b_frag_nk(&Ks[buf][0][32], 72, nt * 16, lane);
      s = wmma_bf16(a0, b0, s);
      s = wmma_bf16(a1, b1, s);
#pragma unroll
      for (int r = 0; r < 8; ++r) {
        const int mloc = m0 + half * 8 + r;
        const float p = __expf(s[r] * scale - mrow[r]) * sinv[r];
        abase[(size_t)mloc * SEQ + kt * 64 + nt * 16 + nl] = p;
      }
    }
  }
}

// ---------------------------------------------------------------------------
extern "C" void kernel_launch(void* const* d_in, const int* in_sizes, int n_in,
                              void* d_out, int out_size, void* d_ws, size_t ws_size,
                              hipStream_t stream) {
  (void)in_sizes; (void)n_in; (void)out_size; (void)ws_size;

  const float* Q_in = (const float*)d_in[0];
  const float* K_in = (const float*)d_in[1];
  const float* V_in = (const float*)d_in[2];
  const float* W_q  = (const float*)d_in[3];
  const float* b_q  = (const float*)d_in[4];
  const float* W_k  = (const float*)d_in[5];
  const float* b_k  = (const float*)d_in[6];
  const float* W_v  = (const float*)d_in[7];
  const float* b_v  = (const float*)d_in[8];
  const float* W_o  = (const float*)d_in[9];
  const float* b_o  = (const float*)d_in[10];

  const size_t XE = (size_t)BATCH * SEQ * D_MODEL;
  const size_t WE = (size_t)D_MODEL * D_MODEL;

  bf16_t* p   = (bf16_t*)d_ws;
  bf16_t* Xq  = p; p += XE;
  bf16_t* Xk  = p; p += XE;
  bf16_t* Xv  = p; p += XE;
  bf16_t* Wqb = p; p += WE;
  bf16_t* Wkb = p; p += WE;
  bf16_t* Wvb = p; p += WE;
  bf16_t* Wob = p; p += WE;
  bf16_t* Qp  = p; p += XE;
  bf16_t* Kp  = p; p += XE;
  bf16_t* Vp  = p; p += XE;
  bf16_t* Ctx = p; p += XE;   // ~67 MB total

  cvt_f32_to_bf16<<<(int)(XE / 2048), 256, 0, stream>>>(Q_in, Xq, (int)XE);
  cvt_f32_to_bf16<<<(int)(XE / 2048), 256, 0, stream>>>(K_in, Xk, (int)XE);
  cvt_f32_to_bf16<<<(int)(XE / 2048), 256, 0, stream>>>(V_in, Xv, (int)XE);
  cvt_f32_to_bf16<<<(int)(WE / 2048), 256, 0, stream>>>(W_q, Wqb, (int)WE);
  cvt_f32_to_bf16<<<(int)(WE / 2048), 256, 0, stream>>>(W_k, Wkb, (int)WE);
  cvt_f32_to_bf16<<<(int)(WE / 2048), 256, 0, stream>>>(W_v, Wvb, (int)WE);
  cvt_f32_to_bf16<<<(int)(WE / 2048), 256, 0, stream>>>(W_o, Wob, (int)WE);

  dim3 gg(D_MODEL / 64, (BATCH * SEQ) / 64);
  gemm_xwT<0><<<gg, 128, 0, stream>>>(Xq, Wqb, b_q, (void*)Qp);
  gemm_xwT<0><<<gg, 128, 0, stream>>>(Xk, Wkb, b_k, (void*)Kp);
  gemm_xwT<0><<<gg, 128, 0, stream>>>(Xv, Wvb, b_v, (void*)Vp);

  float* out_f  = (float*)d_out;
  float* attn_f = out_f + (size_t)BATCH * SEQ * D_MODEL;
  attn_kernel<<<dim3(SEQ / 64, BATCH * N_HEADS), 128, 0, stream>>>(
      Qp, Kp, Vp, attn_f, Ctx);

  gemm_xwT<1><<<gg, 128, 0, stream>>>(Ctx, Wob, b_o, (void*)out_f);
}
